// SurfaceNet_163208757883
// MI455X (gfx1250) — matrix operand order repaired and checked by reading, not compiled, not run on hardware
//
#include <hip/hip_runtime.h>
#include <hip/hip_bf16.h>

// ---------------------------------------------------------------------------
// Frustum PointNet (3->32->256->256 MLP + masked max-pool) for gfx1250.
// Layers 2/3 run on v_wmma_f32_16x16x32_f16 (f16 in, f32 accumulate).
// ---------------------------------------------------------------------------

typedef _Float16 half8 __attribute__((ext_vector_type(8)));
typedef _Float16 v16h  __attribute__((ext_vector_type(16)));
typedef float    v8f   __attribute__((ext_vector_type(8)));

#define NVOX (2 * 96 * 96)   // 18432 voxels
#define FOUT 256
#define NEGF (-1e30f)

#define WPB  2               // waves (= voxels) per block
#define H1S  40              // H1 row stride in halfs (padded, 80B: 16B-aligned runs)
#define H2S  264             // H2 row stride in halfs (padded, 528B: 16B-aligned runs)
#define H1B  (32 * H1S * 2)  // 2560 B
#define H2B  (32 * H2S * 2)  // 16896 B
#define WLDS (H1B + H2B)     // 19456 B per wave

__device__ __forceinline__ v8f splat8(float x) {
  v8f r;
#pragma unroll
  for (int i = 0; i < 8; ++i) r[i] = x;
  return r;
}

__device__ __forceinline__ v16h pack16(half8 lo, half8 hiv) {
  v16h r;
#pragma unroll
  for (int i = 0; i < 8; ++i) { r[i] = lo[i]; r[8 + i] = hiv[i]; }
  return r;
}

// A fragment (16x32 f16, ISA layout): lane holds K runs [k0, k0+8) and
// [k0+16, k0+24) of its row; p points at the first run (16B aligned).
__device__ __forceinline__ v16h load_afrag(const _Float16* p) {
  half8 x0 = *(const half8*)p;
  half8 x1 = *(const half8*)(p + 16);
  return pack16(x0, x1);
}

// B fragment (32x16 f16): lane holds K runs [k0, k0+8) and [k0+16, k0+24) of column n.
__device__ __forceinline__ v16h load_bfrag(const _Float16* p) {
  half8 x0 = *(const half8*)p;
  half8 x1 = *(const half8*)(p + 16);
  return pack16(x0, x1);
}

// ---------------------------------------------------------------------------
// Weight pre-conversion f32 -> f16 into workspace (run each launch; cheap).
// ---------------------------------------------------------------------------
__global__ void cvt_w_f16(const float* __restrict__ w2, const float* __restrict__ w3,
                          _Float16* __restrict__ w2h, _Float16* __restrict__ w3h) {
  int i = blockIdx.x * blockDim.x + threadIdx.x;
  if (i < 256 * 32)  w2h[i] = (_Float16)w2[i];
  if (i < 256 * 256) w3h[i] = (_Float16)w3[i];
}

// ---------------------------------------------------------------------------
// Main kernel: one voxel per wave32.
// ---------------------------------------------------------------------------
__global__ void __launch_bounds__(WPB * 32)
frustum_pointnet(const float* __restrict__ fv, const int* __restrict__ fnum,
                 const float* __restrict__ w1, const float* __restrict__ b1,
                 const float* __restrict__ w2, const float* __restrict__ b2,
                 const float* __restrict__ w3, const float* __restrict__ b3,
                 const _Float16* __restrict__ w2h, const _Float16* __restrict__ w3h,
                 float* __restrict__ out) {
  extern __shared__ char smem[];
  const int lane = threadIdx.x & 31;
  const int wid  = threadIdx.x >> 5;
  const int vox  = blockIdx.x * WPB + wid;
  const int m    = lane & 15;   // row (A) / column (B,C,D) within tile
  const int hi   = lane >> 4;   // lane-half selector

  _Float16* h1 = (_Float16*)(smem + wid * WLDS);
  _Float16* h2 = (_Float16*)(smem + wid * WLDS + H1B);

  const int num = fnum[vox];    // wave-uniform -> uniform branches, EXEC stays all-1s

  if (num > 0) {
    // ---------------- Layer 1: per-lane point, 3 -> 32, f32 FMA ----------------
    const float* pt = fv + ((size_t)vox * 32 + lane) * 3;
    const float px = pt[0], py = pt[1], pz = pt[2];
#pragma unroll
    for (int c = 0; c < 32; ++c) {
      float h = fmaf(w1[c * 3 + 2], pz,
                fmaf(w1[c * 3 + 1], py,
                fmaf(w1[c * 3 + 0], px, b1[c])));
      h1[lane * H1S + c] = (_Float16)fmaxf(h, 0.f);
    }
    __builtin_amdgcn_wave_barrier();  // LDS is wave-private; DS ops are in-order

    // ---------------- Layer 2: [32x32] x [32x256], WMMA, K = 32 ----------------
    {
      v16h a0 = load_afrag(h1 + m * H1S + hi * 8);          // rows 0..15
      v16h a1 = load_afrag(h1 + (16 + m) * H1S + hi * 8);   // rows 16..31
#pragma unroll 1
      for (int g = 0; g < 4; ++g) {                         // 4 groups x 64 cols
        v8f acc0[4], acc1[4];
#pragma unroll
        for (int j = 0; j < 4; ++j) {
          float bias = b2[g * 64 + j * 16 + m];
          acc0[j] = splat8(bias);
          acc1[j] = splat8(bias);
        }
#pragma unroll
        for (int j = 0; j < 4; ++j) {
          int n = g * 64 + j * 16 + m;
          v16h bf = load_bfrag(w2h + n * 32 + hi * 8);
          acc0[j] = __builtin_amdgcn_wmma_f32_16x16x32_f16(false, a0, false, bf,
                                                           (short)0, acc0[j], false, false);
          acc1[j] = __builtin_amdgcn_wmma_f32_16x16x32_f16(false, a1, false, bf,
                                                           (short)0, acc1[j], false, false);
        }
        // ReLU + stash as f16 in LDS (C/D layout: lane = col, vgpr v = row hi*8+v)
#pragma unroll
        for (int j = 0; j < 4; ++j) {
          int n = g * 64 + j * 16 + m;
#pragma unroll
          for (int v = 0; v < 8; ++v) {
            h2[(hi * 8 + v) * H2S + n]        = (_Float16)fmaxf(acc0[j][v], 0.f);
            h2[(16 + hi * 8 + v) * H2S + n]   = (_Float16)fmaxf(acc1[j][v], 0.f);
          }
        }
      }
    }
    __builtin_amdgcn_wave_barrier();

    // -------- Layer 3: [32x256] x [256x256], WMMA, K = 256, + max-pool --------
#pragma unroll 1
    for (int g = 0; g < 4; ++g) {
      v8f acc0[4], acc1[4];
#pragma unroll
      for (int j = 0; j < 4; ++j) {
        float bias = b3[g * 64 + j * 16 + m];
        acc0[j] = splat8(bias);
        acc1[j] = splat8(bias);
      }
#pragma unroll
      for (int kt = 0; kt < 8; ++kt) {
        const int kbase = kt * 32;
        v16h a0 = load_afrag(h2 + m * H2S + kbase + hi * 8);
        v16h a1 = load_afrag(h2 + (16 + m) * H2S + kbase + hi * 8);
#pragma unroll
        for (int j = 0; j < 4; ++j) {
          int n = g * 64 + j * 16 + m;
          v16h bf = load_bfrag(w3h + (size_t)n * 256 + kbase + hi * 8);
          acc0[j] = __builtin_amdgcn_wmma_f32_16x16x32_f16(false, a0, false, bf,
                                                           (short)0, acc0[j], false, false);
          acc1[j] = __builtin_amdgcn_wmma_f32_16x16x32_f16(false, a1, false, bf,
                                                           (short)0, acc1[j], false, false);
        }
      }
      // Masked column-max over the 32 points, directly on the D layout.
#pragma unroll
      for (int j = 0; j < 4; ++j) {
        float cm = NEGF;
#pragma unroll
        for (int v = 0; v < 8; ++v) {
          int p0 = hi * 8 + v;        // rows 0..15 tile
          int p1 = 16 + hi * 8 + v;   // rows 16..31 tile
          cm = fmaxf(cm, (p0 < num) ? acc0[j][v] : NEGF);
          cm = fmaxf(cm, (p1 < num) ? acc1[j][v] : NEGF);
        }
        cm = fmaxf(cm, __shfl_xor(cm, 16, 32));  // combine row halves (same column)
        if (hi == 0) out[(size_t)vox * FOUT + g * 64 + j * 16 + m] = cm;
      }
    }
  } else {
    // ------------- Empty voxel: PointNet of a single all-zero point -------------
    float* h2z = (float*)h1;  // reuse this wave's private LDS (256 floats)
    float h1z[32];
#pragma unroll
    for (int i = 0; i < 32; ++i) h1z[i] = fmaxf(b1[i], 0.f);
#pragma unroll 1
    for (int t = 0; t < 8; ++t) {
      int c = t * 32 + lane;
      float a = b2[c];
#pragma unroll
      for (int i = 0; i < 32; ++i) a = fmaf(w2[c * 32 + i], h1z[i], a);
      h2z[c] = fmaxf(a, 0.f);
    }
    __builtin_amdgcn_wave_barrier();
#pragma unroll 1
    for (int t = 0; t < 8; ++t) {
      int c = t * 32 + lane;
      float a = b3[c];
      for (int i = 0; i < 256; ++i) a = fmaf(w3[c * 256 + i], h2z[i], a);
      out[(size_t)vox * FOUT + c] = a;
    }
  }
}

// ---------------------------------------------------------------------------
extern "C" void kernel_launch(void* const* d_in, const int* in_sizes, int n_in,
                              void* d_out, int out_size, void* d_ws, size_t ws_size,
                              hipStream_t stream) {
  const float* fv   = (const float*)d_in[0];
  const int*   fnum = (const int*)d_in[1];
  const float* w1   = (const float*)d_in[2];
  const float* b1   = (const float*)d_in[3];
  const float* w2   = (const float*)d_in[4];
  const float* b2   = (const float*)d_in[5];
  const float* w3   = (const float*)d_in[6];
  const float* b3   = (const float*)d_in[7];
  float* out = (float*)d_out;

  _Float16* w2h = (_Float16*)d_ws;                           // 256*32  halfs
  _Float16* w3h = (_Float16*)((char*)d_ws + 256 * 32 * 2);   // 256*256 halfs

  // Convert weights to f16 once per launch (deterministic, ordered on stream).
  cvt_w_f16<<<dim3((256 * 256 + 255) / 256), dim3(256), 0, stream>>>(w2, w3, w2h, w3h);

  dim3 grid(NVOX / WPB);     // 9216 blocks
  dim3 block(WPB * 32);      // 2 wave32 per block
  frustum_pointnet<<<grid, block, WPB * WLDS, stream>>>(
      fv, fnum, w1, b1, w2, b2, w3, b3, w2h, w3h, out);
}
